// SSMPathway_60713657697113
// MI455X (gfx1250) — matrix operand order repaired and checked by reading, not compiled, not run on hardware
//
#include <hip/hip_runtime.h>
#include <hip/hip_bf16.h>
#include <math.h>

typedef __attribute__((ext_vector_type(16))) __bf16 v16bf;
typedef __attribute__((ext_vector_type(8)))  __bf16 v8bf;
typedef __attribute__((ext_vector_type(8)))  float  v8f;

// ---------------------------------------------------------------------------
// gfx1250 async global->LDS copy path (ASYNCcnt-tracked), guarded so the
// register-staged synchronous path remains a compile-safe fallback.
// Builtin prototype (from hipcc diagnostic): params are generic pointers to
// 'int __attribute__((vector_size(16)))'.
// ---------------------------------------------------------------------------
#if defined(__gfx1250__) &&                                                    \
    __has_builtin(__builtin_amdgcn_global_load_async_to_lds_b128) &&           \
    __has_builtin(__builtin_amdgcn_s_wait_asynccnt)
#define USE_ASYNC_LDS 1
typedef int async_b128 __attribute__((vector_size(4 * sizeof(int))));
__device__ __forceinline__ void async_cp16(void* l, const void* g) {
    __builtin_amdgcn_global_load_async_to_lds_b128((async_b128*)g, (async_b128*)l, 0, 0);
}
__device__ __forceinline__ void async_wait0() {
    __builtin_amdgcn_s_wait_asynccnt(0);
}
#else
#define USE_ASYNC_LDS 0
#endif

// ---------------------------------------------------------------------------
// fp32 -> bf16 conversion
// ---------------------------------------------------------------------------
__global__ __launch_bounds__(256) void cvt_bf16_kernel(const float* __restrict__ in,
                                                       __bf16* __restrict__ out,
                                                       size_t n) {
    size_t i = (size_t)blockIdx.x * 256 + threadIdx.x;
    if (i < n) out[i] = (__bf16)in[i];
}

// ---------------------------------------------------------------------------
// LayerNorm over H=1024; one block (256 thr) per row, 4 elems/thread
// ---------------------------------------------------------------------------
__global__ __launch_bounds__(256) void ln_kernel(const float* __restrict__ x,
                                                 const float* __restrict__ g,
                                                 const float* __restrict__ b,
                                                 float* __restrict__ xn, int H) {
    __shared__ float red[8];
    const size_t row = blockIdx.x;
    const float* xr = x + row * H;
    int tid = threadIdx.x, lane = tid & 31, wid = tid >> 5;

    float v[4];
    float s = 0.f;
#pragma unroll
    for (int i = 0; i < 4; ++i) { v[i] = xr[tid + i * 256]; s += v[i]; }
#pragma unroll
    for (int o = 16; o > 0; o >>= 1) s += __shfl_xor(s, o, 32);
    if (lane == 0) red[wid] = s;
    __syncthreads();
    float tot = 0.f;
#pragma unroll
    for (int i = 0; i < 8; ++i) tot += red[i];
    __syncthreads();
    float mean = tot * (1.0f / 1024.0f);

    float vs = 0.f;
#pragma unroll
    for (int i = 0; i < 4; ++i) { float d = v[i] - mean; vs += d * d; }
#pragma unroll
    for (int o = 16; o > 0; o >>= 1) vs += __shfl_xor(vs, o, 32);
    if (lane == 0) red[wid] = vs;
    __syncthreads();
    float vtot = 0.f;
#pragma unroll
    for (int i = 0; i < 8; ++i) vtot += red[i];
    float inv = rsqrtf(vtot * (1.0f / 1024.0f) + 1e-5f);

    float* xo = xn + row * H;
#pragma unroll
    for (int i = 0; i < 4; ++i) {
        int h = tid + i * 256;
        xo[h] = (v[i] - mean) * inv * g[h] + b[h];
    }
}

// ---------------------------------------------------------------------------
// Depthwise conv (K=3, pad=1) + residual + bias; emits fp32 and bf16 copies
// ---------------------------------------------------------------------------
__global__ __launch_bounds__(256) void conv_kernel(const float* __restrict__ xn,
                                                   const float* __restrict__ cw,
                                                   const float* __restrict__ cb,
                                                   float* __restrict__ xn2,
                                                   __bf16* __restrict__ xnb,
                                                   int T, int H) {
    size_t idx = (size_t)blockIdx.x * 256 + threadIdx.x; // over B*T*H
    int h = (int)(idx % H);
    size_t bt = idx / H;
    int t = (int)(bt % T);
    float w0 = cw[h * 3 + 0], w1 = cw[h * 3 + 1], w2 = cw[h * 3 + 2];
    float acc = xn[idx] + cb[h] + w1 * xn[idx];
    if (t > 0)      acc += w0 * xn[idx - H];
    if (t < T - 1)  acc += w2 * xn[idx + H];
    xn2[idx] = acc;
    xnb[idx] = (__bf16)acc;
}

// ---------------------------------------------------------------------------
// expm(dt*A) for 64x64 via Taylor series (||dt*A|| ~ 0.1 -> 12 terms ample)
// ---------------------------------------------------------------------------
__global__ __launch_bounds__(256) void expm_kernel(const float* __restrict__ A,
                                                   const float* __restrict__ dtp,
                                                   float* __restrict__ Abar) {
    __shared__ float Mx[64 * 64];
    __shared__ float P0[64 * 64];
    __shared__ float P1[64 * 64];
    float dt = dtp[0];
    int tid = threadIdx.x;
    for (int i = tid; i < 4096; i += 256) {
        float m = dt * A[i];
        Mx[i] = m; P0[i] = m;
    }
    __syncthreads();
    int r = tid >> 2, c0 = (tid & 3) * 16;
    float E[16];
#pragma unroll
    for (int i = 0; i < 16; ++i) {
        int c = c0 + i;
        E[i] = (r == c ? 1.0f : 0.0f) + Mx[r * 64 + c];
    }
    float* Pa = P0; float* Pb = P1;
    for (int n = 2; n <= 12; ++n) {
        float invn = 1.0f / (float)n;
#pragma unroll
        for (int i = 0; i < 16; ++i) {
            int c = c0 + i;
            float s = 0.f;
            for (int k = 0; k < 64; ++k) s += Pa[r * 64 + k] * Mx[k * 64 + c];
            float pv = s * invn;
            Pb[r * 64 + c] = pv;
            E[i] += pv;
        }
        __syncthreads();
        float* tsw = Pa; Pa = Pb; Pb = tsw;
    }
#pragma unroll
    for (int i = 0; i < 16; ++i) Abar[r * 64 + c0 + i] = E[i];
}

// ---------------------------------------------------------------------------
// WMMA bf16 GEMM: C[M,N] = A[M,K] @ W[N,K]^T, fp32 accum, templated epilogue.
// Block: 256 thr = 8 waves arranged WGM x WGN; wave tile = SUBM*16 x SUBN*16.
// Double-buffered LDS, one barrier per K-step, async global->LDS when present.
// Fragment gathers follow CDNA5 ISA 16-bit A(16x32)/B(32x16) layouts.
// ---------------------------------------------------------------------------
__device__ __forceinline__ v16bf load_frag16(const __bf16* p0, const __bf16* p1) {
    v8bf x = *(const v8bf*)p0;
    v8bf y = *(const v8bf*)p1;
    v16bf r;
#pragma unroll
    for (int i = 0; i < 8; ++i) { r[i] = x[i]; r[i + 8] = y[i]; }
    return r;
}

template <int WGM, int WGN, int SUBM, int SUBN, typename Epi>
__global__ __launch_bounds__(256) void wmma_gemm_kernel(const __bf16* __restrict__ A,
                                                        const __bf16* __restrict__ W,
                                                        int M, int N, int K, Epi epi) {
    constexpr int BM  = WGM * SUBM * 16;
    constexpr int BN  = WGN * SUBN * 16;
    constexpr int ACH = (BM * 32) / (256 * 8); // 8-elem chunks per thread (A tile)
    constexpr int BCH = (BN * 32) / (256 * 8); // 8-elem chunks per thread (B tile)
    __shared__ __bf16 As[2][BM * 32];
    __shared__ __bf16 Bs[2][BN * 32];

    const int tid  = threadIdx.x;
    const int bm   = blockIdx.y * BM, bn = blockIdx.x * BN;
    const int wave = tid >> 5, lane = tid & 31;
    const int wn   = wave % WGN, wm = wave / WGN;
    const int l15  = lane & 15;
    const int akoff = (lane >> 4) * 8;   // A: lane<16 -> K 0..7/16..23, else 8..15/24..31
    const int bkoff = (lane >> 4) * 8;   // B: lane<16 -> K 0..7/16..23, else 8..15/24..31

    v8f acc[SUBM][SUBN];
    v8f zero = {};
#pragma unroll
    for (int s = 0; s < SUBM; ++s)
#pragma unroll
        for (int u = 0; u < SUBN; ++u) acc[s][u] = zero;

    // ---- prologue: stage K-tile 0 into buffer 0
#if USE_ASYNC_LDS
#pragma unroll
    for (int i = 0; i < ACH; ++i) {
        int ch = tid + i * 256;
        async_cp16(&As[0][(ch >> 2) * 32 + (ch & 3) * 8],
                   A + (size_t)(bm + (ch >> 2)) * K + (ch & 3) * 8);
    }
#pragma unroll
    for (int i = 0; i < BCH; ++i) {
        int ch = tid + i * 256;
        async_cp16(&Bs[0][(ch >> 2) * 32 + (ch & 3) * 8],
                   W + (size_t)(bn + (ch >> 2)) * K + (ch & 3) * 8);
    }
#else
    {
        v8bf ra[ACH], rb[BCH];
#pragma unroll
        for (int i = 0; i < ACH; ++i) {
            int ch = tid + i * 256;
            ra[i] = *(const v8bf*)(A + (size_t)(bm + (ch >> 2)) * K + (ch & 3) * 8);
        }
#pragma unroll
        for (int i = 0; i < BCH; ++i) {
            int ch = tid + i * 256;
            rb[i] = *(const v8bf*)(W + (size_t)(bn + (ch >> 2)) * K + (ch & 3) * 8);
        }
#pragma unroll
        for (int i = 0; i < ACH; ++i) {
            int ch = tid + i * 256;
            *(v8bf*)&As[0][(ch >> 2) * 32 + (ch & 3) * 8] = ra[i];
        }
#pragma unroll
        for (int i = 0; i < BCH; ++i) {
            int ch = tid + i * 256;
            *(v8bf*)&Bs[0][(ch >> 2) * 32 + (ch & 3) * 8] = rb[i];
        }
    }
#endif

    int buf = 0;
    for (int k0 = 0; k0 < K; k0 += 32) {
#if USE_ASYNC_LDS
        async_wait0();     // my async stores into As/Bs[buf] are complete
#endif
        __syncthreads();   // everyone's tile visible; prior reads of buf^1 done
        const bool next = (k0 + 32) < K;

#if USE_ASYNC_LDS
        if (next) {
#pragma unroll
            for (int i = 0; i < ACH; ++i) {
                int ch = tid + i * 256;
                async_cp16(&As[buf ^ 1][(ch >> 2) * 32 + (ch & 3) * 8],
                           A + (size_t)(bm + (ch >> 2)) * K + k0 + 32 + (ch & 3) * 8);
            }
#pragma unroll
            for (int i = 0; i < BCH; ++i) {
                int ch = tid + i * 256;
                async_cp16(&Bs[buf ^ 1][(ch >> 2) * 32 + (ch & 3) * 8],
                           W + (size_t)(bn + (ch >> 2)) * K + k0 + 32 + (ch & 3) * 8);
            }
        }
#else
        v8bf ra[ACH], rb[BCH];
        if (next) {
#pragma unroll
            for (int i = 0; i < ACH; ++i) {
                int ch = tid + i * 256;
                ra[i] = *(const v8bf*)(A + (size_t)(bm + (ch >> 2)) * K + k0 + 32 + (ch & 3) * 8);
            }
#pragma unroll
            for (int i = 0; i < BCH; ++i) {
                int ch = tid + i * 256;
                rb[i] = *(const v8bf*)(W + (size_t)(bn + (ch >> 2)) * K + k0 + 32 + (ch & 3) * 8);
            }
            __builtin_prefetch(A + (size_t)(bm + (tid >> 2)) * K + k0 + 64, 0, 1);
        }
#endif

        // ---- compute from As/Bs[buf]
        {
            const __bf16* Ab = As[buf];
            const __bf16* Bb = Bs[buf];
            v16bf af[SUBM], bfv[SUBN];
#pragma unroll
            for (int s = 0; s < SUBM; ++s) {
                int row = (wm * SUBM + s) * 16 + l15;
                af[s] = load_frag16(&Ab[row * 32 + akoff], &Ab[row * 32 + 16 + akoff]);
            }
#pragma unroll
            for (int u = 0; u < SUBN; ++u) {
                int row = (wn * SUBN + u) * 16 + l15;
                bfv[u] = load_frag16(&Bb[row * 32 + bkoff], &Bb[row * 32 + bkoff + 16]);
            }
#pragma unroll
            for (int s = 0; s < SUBM; ++s)
#pragma unroll
                for (int u = 0; u < SUBN; ++u)
                    acc[s][u] = __builtin_amdgcn_wmma_f32_16x16x32_bf16(
                        false, af[s], false, bfv[u], (short)0, acc[s][u], false, false);
        }

#if !USE_ASYNC_LDS
        if (next) {
#pragma unroll
            for (int i = 0; i < ACH; ++i) {
                int ch = tid + i * 256;
                *(v8bf*)&As[buf ^ 1][(ch >> 2) * 32 + (ch & 3) * 8] = ra[i];
            }
#pragma unroll
            for (int i = 0; i < BCH; ++i) {
                int ch = tid + i * 256;
                *(v8bf*)&Bs[buf ^ 1][(ch >> 2) * 32 + (ch & 3) * 8] = rb[i];
            }
        }
#endif
        buf ^= 1;
    }

    // C/D layout: lane L, vgpr r -> M = r + (L>>4)*8, N = L&15
#pragma unroll
    for (int s = 0; s < SUBM; ++s) {
        int m0 = bm + (wm * SUBM + s) * 16 + ((lane >> 4) * 8);
#pragma unroll
        for (int u = 0; u < SUBN; ++u) {
            int n = bn + (wn * SUBN + u) * 16 + l15;
#pragma unroll
            for (int rr = 0; rr < 8; ++rr) epi(m0 + rr, n, acc[s][u][rr]);
        }
    }
}

// epilogues ------------------------------------------------------------------
struct GateEpi {
    const float* bias; float* gate; __bf16* xg; const __bf16* xn; int H;
    __device__ void operator()(int m, int n, float v) const {
        float t = v + bias[n];
        float s = 1.0f / (1.0f + __expf(-t));
        if (n < H) {
            gate[(size_t)m * H + n] = s;
        } else {
            size_t i = (size_t)m * H + (n - H);
            xg[i] = (__bf16)((float)xn[i] * s);
        }
    }
};
struct BiasEpi {
    const float* bias; float* out; int N;
    __device__ void operator()(int m, int n, float v) const {
        out[(size_t)m * N + n] = v + bias[n];
    }
};
struct DGateEpi {
    const float* bias; const float* gate; float* out; int N;
    __device__ void operator()(int m, int n, float v) const {
        size_t i = (size_t)m * N + n;
        out[i] = (v + bias[n]) * gate[i];
    }
};

// ---------------------------------------------------------------------------
// Sequential scan: s_t = LN(s_{t-1} @ A_bar + b_t); y = c_t * s_t
// One block, 256 threads: thread = (b = tid/64, j = tid%64).
// ---------------------------------------------------------------------------
__global__ __launch_bounds__(256) void scan_kernel(const float* __restrict__ Abar,
                                                   const float* __restrict__ Bm,
                                                   const float* __restrict__ Cm,
                                                   const float* __restrict__ g,
                                                   const float* __restrict__ bb,
                                                   __bf16* __restrict__ ssm, int T) {
    __shared__ float Ab[64 * 64];
    __shared__ float st[256];
    __shared__ float redS[8];
    __shared__ float redQ[8];
    const int tid = threadIdx.x, lane = tid & 31, wid = tid >> 5;
    const int b = tid >> 6, j = tid & 63;
    for (int i = tid; i < 4096; i += 256) Ab[i] = Abar[i];
    st[tid] = 0.0f;
    __syncthreads();
    const float gj = g[j], bj = bb[j];
    const float* srow = &st[b * 64];

    for (int t = 0; t < T; ++t) {
        size_t base = ((size_t)b * T + t) * 64 + j;
        float v = Bm[base];
        float ct = Cm[base];
#pragma unroll 8
        for (int k = 0; k < 64; ++k) v += srow[k] * Ab[k * 64 + j];

        float s1 = v, s2 = v * v;
#pragma unroll
        for (int o = 16; o > 0; o >>= 1) {
            s1 += __shfl_xor(s1, o, 32);
            s2 += __shfl_xor(s2, o, 32);
        }
        if (lane == 0) { redS[wid] = s1; redQ[wid] = s2; }
        __syncthreads();
        float mean = (redS[b * 2] + redS[b * 2 + 1]) * (1.0f / 64.0f);
        float msq  = (redQ[b * 2] + redQ[b * 2 + 1]) * (1.0f / 64.0f);
        float var  = msq - mean * mean;
        float sn   = (v - mean) * rsqrtf(var + 1e-5f) * gj + bj;
        st[tid] = sn;
        ssm[base] = (__bf16)(ct * sn);
        __syncthreads();
    }
}

// ---------------------------------------------------------------------------
extern "C" void kernel_launch(void* const* d_in, const int* in_sizes, int n_in,
                              void* d_out, int out_size, void* d_ws, size_t ws_size,
                              hipStream_t stream) {
    const float* x       = (const float*)d_in[0];
    const float* A       = (const float*)d_in[1];
    const float* dtA     = (const float*)d_in[2];
    const float* B_w     = (const float*)d_in[3];
    const float* B_b     = (const float*)d_in[4];
    const float* C_w     = (const float*)d_in[5];
    const float* C_b     = (const float*)d_in[6];
    const float* D_w     = (const float*)d_in[7];
    const float* D_b     = (const float*)d_in[8];
    const float* conv_w  = (const float*)d_in[9];
    const float* conv_b  = (const float*)d_in[10];
    const float* gate_w  = (const float*)d_in[11];
    const float* gate_b  = (const float*)d_in[12];
    const float* ln_in_g = (const float*)d_in[13];
    const float* ln_in_b = (const float*)d_in[14];
    const float* ln_st_g = (const float*)d_in[15];
    const float* ln_st_b = (const float*)d_in[16];

    constexpr int    Bt = 4, T = 4096, H = 1024, S = 64;
    constexpr size_t MT = (size_t)Bt * T; // 16384 rows

    char* p = (char*)d_ws;
    auto alloc = [&](size_t bytes) -> char* {
        char* r = p;
        p += (bytes + 255) & ~(size_t)255;
        return r;
    };
    float*  xn   = (float*) alloc(MT * H * 4);
    float*  xn2  = (float*) alloc(MT * H * 4);
    __bf16* xnb  = (__bf16*)alloc(MT * H * 2);
    __bf16* xgb  = (__bf16*)alloc(MT * H * 2);
    float*  gate = (float*) alloc(MT * H * 4);
    float*  Bm   = (float*) alloc(MT * S * 4);
    float*  Cm   = (float*) alloc(MT * S * 4);
    __bf16* ssm  = (__bf16*)alloc(MT * S * 2);
    float*  Abar = (float*) alloc((size_t)S * S * 4);
    __bf16* gwb  = (__bf16*)alloc((size_t)2 * H * H * 2);
    __bf16* Bwb  = (__bf16*)alloc((size_t)S * H * 2);
    __bf16* Cwb  = (__bf16*)alloc((size_t)S * H * 2);
    __bf16* Dwb  = (__bf16*)alloc((size_t)H * S * 2);
    (void)xn2; (void)ws_size; (void)in_sizes; (void)n_in; (void)out_size;

    // weight precision: bf16 operands, fp32 accumulate (WMMA path)
    cvt_bf16_kernel<<<(2 * H * H + 255) / 256, 256, 0, stream>>>(gate_w, gwb, (size_t)2 * H * H);
    cvt_bf16_kernel<<<(S * H + 255) / 256, 256, 0, stream>>>(B_w, Bwb, (size_t)S * H);
    cvt_bf16_kernel<<<(S * H + 255) / 256, 256, 0, stream>>>(C_w, Cwb, (size_t)S * H);
    cvt_bf16_kernel<<<(H * S + 255) / 256, 256, 0, stream>>>(D_w, Dwb, (size_t)H * S);

    ln_kernel<<<(unsigned)MT, 256, 0, stream>>>(x, ln_in_g, ln_in_b, xn, H);
    conv_kernel<<<(unsigned)(MT * H / 256), 256, 0, stream>>>(xn, conv_w, conv_b, xn2, xnb, T, H);
    expm_kernel<<<1, 256, 0, stream>>>(A, dtA, Abar);

    // gate GEMM: [MT x 2H] = xnb @ gwb^T, fused sigmoid -> gate, xg
    // config: BM=128, BN=128 (waves 2x4, wave tile 64x32)
    wmma_gemm_kernel<2, 4, 4, 2><<<dim3(2 * H / 128, (unsigned)(MT / 128)), 256, 0, stream>>>(
        xnb, gwb, (int)MT, 2 * H, H, GateEpi{gate_b, gate, xgb, xnb, H});
    // Bm = xg @ B_w^T + B_b ; Cm = xn @ C_w^T + C_b
    // config: BM=128, BN=64 (waves 4x2, wave tile 32x32)
    wmma_gemm_kernel<4, 2, 2, 2><<<dim3(1, (unsigned)(MT / 128)), 256, 0, stream>>>(
        xgb, Bwb, (int)MT, S, H, BiasEpi{B_b, Bm, S});
    wmma_gemm_kernel<4, 2, 2, 2><<<dim3(1, (unsigned)(MT / 128)), 256, 0, stream>>>(
        xnb, Cwb, (int)MT, S, H, BiasEpi{C_b, Cm, S});

    scan_kernel<<<1, 256, 0, stream>>>(Abar, Bm, Cm, ln_st_g, ln_st_b, ssm, T);

    // out = (ssm @ D_w^T + D_b) * gate, K=64
    wmma_gemm_kernel<2, 4, 4, 2><<<dim3(H / 128, (unsigned)(MT / 128)), 256, 0, stream>>>(
        ssm, Dwb, (int)MT, H, S, DGateEpi{D_b, gate, (float*)d_out, H});
}
